// CausalSelfAttention_85392539779597
// MI455X (gfx1250) — matrix operand order repaired and checked by reading, not compiled, not run on hardware
//
#include <hip/hip_runtime.h>

// MI455X / gfx1250 CausalSelfAttention:
//  - all matmuls on v_wmma_f32_16x16x32_f16 (f32 accumulate)
//  - GEMM B-tiles staged in LDS via TDM tensor_load_to_lds (double-buffered,
//    padded rows -> per-half column reads, conflict-reduced)
//  - flash-attention inner loop with LDS P-tile relayout, wave32 throughout

typedef _Float16 half_t;
typedef __attribute__((ext_vector_type(16))) _Float16 v16h;
typedef __attribute__((ext_vector_type(8)))  _Float16 v8h;
typedef __attribute__((ext_vector_type(8)))  float    v8f;
typedef __attribute__((ext_vector_type(4)))  unsigned int u32x4;
typedef __attribute__((ext_vector_type(8)))  int          i32x8;
typedef __attribute__((ext_vector_type(4)))  int          i32x4;

#define WMMA_F16(A, B, C) \
  __builtin_amdgcn_wmma_f32_16x16x32_f16(false, (A), false, (B), (short)0, (C), false, false)

#if __has_builtin(__builtin_amdgcn_tensor_load_to_lds)
#define USE_TDM 1
#else
#define USE_TDM 0
#endif

static constexpr int B_  = 2;
static constexpr int T_  = 2048;
static constexpr int C_  = 1024;
static constexpr int H_  = 16;
static constexpr int D_  = 64;
static constexpr int BT_ = B_ * T_;   // 4096

// LDS B-tile: 32 k-rows x 64 n-cols f16, row pitch 72 halves (144B):
// TDM pad inserts 4 dwords per 32-dword row; 16B-aligned rows; B read as per-half column loads.
static constexpr int BROW_ = 72;
static constexpr int BBUF_ = 32 * BROW_;          // halves per buffer (4608 B)

__device__ __forceinline__ v16h ld_frag(const half_t* p0, const half_t* p1) {
  v8h lo = *(const v8h*)p0;
  v8h hi = *(const v8h*)p1;
  return __builtin_shufflevector(lo, hi, 0,1,2,3,4,5,6,7,8,9,10,11,12,13,14,15);
}

__device__ __forceinline__ v16h ld_col(const half_t* p, int ld, int g) {
  v16h r;
  #pragma unroll
  for (int s = 0; s < 16; ++s) {
    int k = (s < 8) ? (8 * g + s) : (16 + 8 * g + (s - 8));
    r[s] = p[(size_t)k * ld];
  }
  return r;
}

#if USE_TDM
// D#: 2D tile load, data_size=2B, tile 64(n) x 32(k), tensor row stride = ldn,
// pad_enable: every 32 dwords (one 128B row) add 4 dwords -> LDS pitch 144B.
__device__ __forceinline__ void tdm_load_tile(unsigned lds_addr, const half_t* gptr,
                                              unsigned ldn, unsigned krows) {
  unsigned long long ga = (unsigned long long)(uintptr_t)gptr;
  u32x4 g0 = {
    1u,                                                  // count=1, user load
    lds_addr,                                            // lds_addr
    (unsigned)ga,                                        // global_addr[31:0]
    (unsigned)((ga >> 32) & 0x1FFFFFFu) | (2u << 30)     // global_addr[56:32] | type=2
  };
  i32x8 g1 = {
    (int)((1u << 16) | (1u << 20) | (4u << 22) | (3u << 25)), // 2B elems, pad_en, ivl=32dw, amt=4dw
    (int)((ldn & 0xFFFFu) << 16),                        // tensor_dim0[15:0]
    (int)((ldn >> 16) | ((krows & 0xFFFFu) << 16)),      // tensor_dim0[31:16] | tensor_dim1[15:0]
    (int)((krows >> 16) | (64u << 16)),                  // tensor_dim1[31:16] | tile_dim0=64
    32,                                                  // tile_dim1=32, tile_dim2=0
    (int)ldn,                                            // tensor_dim0_stride[31:0]
    0, 0
  };
  i32x4 z4 = {0, 0, 0, 0};
  i32x8 z8 = {0, 0, 0, 0, 0, 0, 0, 0};
  __builtin_amdgcn_tensor_load_to_lds(g0, g1, z4, z4, z8, 0);  // 6-arg (clang-23) form
}
#endif

// ---------------- snap_weight support ----------------

__global__ void absmax_kernel(const float* __restrict__ w, int n, unsigned int* __restrict__ out) {
  float m = 0.f;
  int stride = gridDim.x * blockDim.x;
  for (int i = blockIdx.x * blockDim.x + threadIdx.x; i < n; i += stride)
    m = fmaxf(m, fabsf(w[i]));
  #pragma unroll
  for (int s = 1; s < 32; s <<= 1) m = fmaxf(m, __shfl_xor(m, s));
  if ((threadIdx.x & 31) == 0) atomicMax(out, __float_as_uint(m));  // w>=0: uint bits ordered
}

__device__ const float SNAP_LUT[16] = {
  0.0f, 1.0f, -1.0f, 0.5f, -0.5f, 0.333333f, -0.333333f, 0.2f, -0.2f,
  0.142857f, -0.142857f, 0.090909f, -0.090909f, 0.076923f, -0.076923f, 0.0f };

__global__ void snap_transpose_kernel(const float* __restrict__ w,
                                      const unsigned int* __restrict__ sc,
                                      half_t* __restrict__ wt, int rows, int cols) {
  int idx = blockIdx.x * blockDim.x + threadIdx.x;
  int n = rows * cols;
  if (idx >= n) return;
  float scale = __uint_as_float(*sc) + 1e-6f;
  int j = idx / cols;
  int k = idx - j * cols;
  float wn = w[idx] / scale;
  float best = SNAP_LUT[0];
  float bd   = fabsf(wn - best);
  #pragma unroll
  for (int t = 1; t < 16; ++t) {
    float d = fabsf(wn - SNAP_LUT[t]);
    if (d < bd) { bd = d; best = SNAP_LUT[t]; }
  }
  wt[(size_t)k * rows + j] = (half_t)(best * scale);
}

__global__ void cvt_half_kernel(const float* __restrict__ x, half_t* __restrict__ xh, int n) {
  int i = blockIdx.x * blockDim.x + threadIdx.x;
  if (i < n) xh[i] = (half_t)x[i];
}

// ---------------- GEMM1: qkv = x @ Wt + b, split/scatter into Q,Kt,V ----------------
// grid(16,48) block 256: block tile 256(M)x64(N); wave -> 32x64; B staged in LDS by TDM.

__global__ __launch_bounds__(256) void qkv_gemm_kernel(
    const half_t* __restrict__ X, const half_t* __restrict__ Wt,
    const float* __restrict__ bias,
    half_t* __restrict__ Q, half_t* __restrict__ Kt, half_t* __restrict__ V) {
  __shared__ half_t bsh[2][BBUF_];
  const int lane = threadIdx.x & 31;
  const int wv   = threadIdx.x >> 5;
  const int m0   = blockIdx.x * 256 + wv * 32;
  const int n0   = blockIdx.y * 64;
  const int g = lane >> 4, lr = lane & 15;
  const int ldn = 3 * C_;
#if USE_TDM
  const unsigned lds_base = (unsigned)(uintptr_t)(&bsh[0][0]);
#endif

  // stage k-tile 0 into buffer 0
#if USE_TDM
  if (wv == 0) {
    tdm_load_tile(lds_base, Wt + n0, (unsigned)ldn, (unsigned)C_);
    __builtin_amdgcn_s_wait_tensorcnt(0);
  }
#else
  {
    int rr = threadIdx.x >> 3, cc = (threadIdx.x & 7) * 8;
    *(v8h*)&bsh[0][rr * BROW_ + cc] = *(const v8h*)(Wt + (size_t)rr * ldn + n0 + cc);
  }
#endif
  __syncthreads();

  v8f acc[2][4] = {};
  const half_t* arow0 = X + (size_t)(m0 + lr) * C_;
  const half_t* arow1 = X + (size_t)(m0 + 16 + lr) * C_;

  for (int k0 = 0; k0 < C_; k0 += 32) {
    const int cur = (k0 >> 5) & 1;
    const int nxt = cur ^ 1;
    if (k0 + 32 < C_) {
#if USE_TDM
      if (wv == 0)
        tdm_load_tile(lds_base + (unsigned)nxt * (BBUF_ * 2),
                      Wt + (size_t)(k0 + 32) * ldn + n0, (unsigned)ldn, (unsigned)C_);
#else
      int rr = threadIdx.x >> 3, cc = (threadIdx.x & 7) * 8;
      *(v8h*)&bsh[nxt][rr * BROW_ + cc] =
          *(const v8h*)(Wt + (size_t)(k0 + 32 + rr) * ldn + n0 + cc);
#endif
    }
    const half_t* pa0 = arow0 + k0 + g * 8;
    const half_t* pa1 = arow1 + k0 + g * 8;
    v16h a0 = ld_frag(pa0, pa0 + 16);
    v16h a1 = ld_frag(pa1, pa1 + 16);
    const half_t* bp = &bsh[cur][lr];      // lr = local n col
    #pragma unroll
    for (int t = 0; t < 4; ++t) {
      v16h b = ld_col(bp + t * 16, BROW_, g);
      acc[0][t] = WMMA_F16(a0, b, acc[0][t]);
      acc[1][t] = WMMA_F16(a1, b, acc[1][t]);
    }
#if USE_TDM
    if (wv == 0 && k0 + 32 < C_) __builtin_amdgcn_s_wait_tensorcnt(0);
#endif
    __syncthreads();
  }

  #pragma unroll
  for (int t = 0; t < 4; ++t) {
    int n = n0 + t * 16 + lr;
    float bv = bias[n];
    int part = n >> 10;        // 0:q 1:k 2:v
    int cn = n & (C_ - 1);
    int h = cn >> 6, dd = cn & 63;
    #pragma unroll
    for (int ms = 0; ms < 2; ++ms) {
      #pragma unroll
      for (int r = 0; r < 8; ++r) {
        int m = m0 + ms * 16 + 8 * g + r;
        int b_ = m >> 11, tt = m & (T_ - 1);
        float val = acc[ms][t][r] + bv;
        size_t bh = (size_t)(b_ * H_ + h);
        if (part == 0)      Q [(bh * T_ + tt) * D_ + dd] = (half_t)(val * 0.125f);
        else if (part == 1) Kt[(bh * D_ + dd) * T_ + tt] = (half_t)val;
        else                V [(bh * T_ + tt) * D_ + dd] = (half_t)val;
      }
    }
  }
}

// ---------------- flash attention: one wave per 16-query tile ----------------

__global__ __launch_bounds__(256) void attn_kernel(
    const half_t* __restrict__ Q, const half_t* __restrict__ Kt,
    const half_t* __restrict__ V, half_t* __restrict__ Y) {
  __shared__ half_t pbuf[8 * 512];            // 16x32 f16 P-tile per wave
  const int lane = threadIdx.x & 31;
  const int wv   = threadIdx.x >> 5;
  const int w    = blockIdx.x * 8 + wv;       // 0..4095
  const int qt   = w & 127;
  const int bh   = w >> 7;                    // 0..31
  const int b = bh >> 4, h = bh & 15;
  const int g = lane >> 4, lr = lane & 15;
  const int qBase = qt * 16;

  const half_t* Qp = Q  + (size_t)bh * T_ * D_;
  const half_t* Kp = Kt + (size_t)bh * D_ * T_;
  const half_t* Vp = V  + (size_t)bh * T_ * D_;
  half_t* pb = pbuf + wv * 512;

  v16h qa[2];
  {
    const half_t* qr = Qp + (size_t)(qBase + lr) * D_;
    #pragma unroll
    for (int kk = 0; kk < 2; ++kk) {
      const half_t* p = qr + kk * 32 + g * 8;
      qa[kk] = ld_frag(p, p + 16);
    }
  }
  v8f acc[4] = {};
  float mrow[8], lrow[8];
  #pragma unroll
  for (int r = 0; r < 8; ++r) { mrow[r] = -__builtin_inff(); lrow[r] = 0.f; }

  for (int kb = 0; kb <= qBase + 15; kb += 32) {      // uniform per wave -> EXEC all-1s
    v8f S[2] = {};
    #pragma unroll
    for (int sub = 0; sub < 2; ++sub) {
      #pragma unroll
      for (int kk = 0; kk < 2; ++kk) {
        const half_t* p = Kp + (size_t)(kk * 32) * T_ + kb + sub * 16 + lr;
        v16h bf = ld_col(p, T_, g);
        S[sub] = WMMA_F16(qa[kk], bf, S[sub]);
      }
    }
    #pragma unroll
    for (int r = 0; r < 8; ++r) {
      const int q = qBase + 8 * g + r;
      #pragma unroll
      for (int sub = 0; sub < 2; ++sub) {
        int key = kb + sub * 16 + lr;
        if (key > q) S[sub][r] = -__builtin_inff();
      }
      float v = fmaxf(S[0][r], S[1][r]);
      #pragma unroll
      for (int msk = 1; msk < 16; msk <<= 1) v = fmaxf(v, __shfl_xor(v, msk));
      float nm = fmaxf(mrow[r], v);
      float alpha = __expf(mrow[r] - nm);
      lrow[r] *= alpha;
      #pragma unroll
      for (int t = 0; t < 4; ++t) acc[t][r] *= alpha;
      float p0 = __expf(S[0][r] - nm);
      float p1 = __expf(S[1][r] - nm);
      S[0][r] = p0; S[1][r] = p1;
      float ps = p0 + p1;
      #pragma unroll
      for (int msk = 1; msk < 16; msk <<= 1) ps += __shfl_xor(ps, msk);
      lrow[r] += ps;
      mrow[r] = nm;
    }
    // P (C-layout) -> LDS -> A-layout
    #pragma unroll
    for (int r = 0; r < 8; ++r) {
      int row = 8 * g + r;
      pb[row * 32 + lr]      = (half_t)S[0][r];
      pb[row * 32 + 16 + lr] = (half_t)S[1][r];
    }
    asm volatile("s_wait_dscnt 0x0" ::: "memory");
    const half_t* pr = pb + lr * 32 + g * 8;
    v16h pa = ld_frag(pr, pr + 16);
    #pragma unroll
    for (int t = 0; t < 4; ++t) {
      const half_t* p = Vp + (size_t)kb * D_ + t * 16 + lr;
      v16h bf = ld_col(p, D_, g);
      acc[t] = WMMA_F16(pa, bf, acc[t]);
    }
  }
  #pragma unroll
  for (int r = 0; r < 8; ++r) {
    float inv = 1.f / lrow[r];
    int q = qBase + 8 * g + r;
    half_t* yrow = Y + ((size_t)(b * T_ + q)) * C_ + h * D_;
    #pragma unroll
    for (int t = 0; t < 4; ++t)
      yrow[t * 16 + lr] = (half_t)(acc[t][r] * inv);
  }
}

// ---------------- GEMM2: out = y @ WtP + b (f32 out) ----------------
// grid(16,16) block 256: block tile 256x64; wave 32x64; B staged via TDM.

__global__ __launch_bounds__(256) void proj_gemm_kernel(
    const half_t* __restrict__ Yh, const half_t* __restrict__ Wt,
    const float* __restrict__ bias, float* __restrict__ out) {
  __shared__ half_t bsh[2][BBUF_];
  const int lane = threadIdx.x & 31;
  const int wv   = threadIdx.x >> 5;
  const int m0   = blockIdx.x * 256 + wv * 32;
  const int n0   = blockIdx.y * 64;
  const int g = lane >> 4, lr = lane & 15;
  const int ldn = C_;
#if USE_TDM
  const unsigned lds_base = (unsigned)(uintptr_t)(&bsh[0][0]);
#endif

#if USE_TDM
  if (wv == 0) {
    tdm_load_tile(lds_base, Wt + n0, (unsigned)ldn, (unsigned)C_);
    __builtin_amdgcn_s_wait_tensorcnt(0);
  }
#else
  {
    int rr = threadIdx.x >> 3, cc = (threadIdx.x & 7) * 8;
    *(v8h*)&bsh[0][rr * BROW_ + cc] = *(const v8h*)(Wt + (size_t)rr * ldn + n0 + cc);
  }
#endif
  __syncthreads();

  v8f acc[2][4] = {};
  const half_t* arow0 = Yh + (size_t)(m0 + lr) * C_;
  const half_t* arow1 = Yh + (size_t)(m0 + 16 + lr) * C_;

  for (int k0 = 0; k0 < C_; k0 += 32) {
    const int cur = (k0 >> 5) & 1;
    const int nxt = cur ^ 1;
    if (k0 + 32 < C_) {
#if USE_TDM
      if (wv == 0)
        tdm_load_tile(lds_base + (unsigned)nxt * (BBUF_ * 2),
                      Wt + (size_t)(k0 + 32) * ldn + n0, (unsigned)ldn, (unsigned)C_);
#else
      int rr = threadIdx.x >> 3, cc = (threadIdx.x & 7) * 8;
      *(v8h*)&bsh[nxt][rr * BROW_ + cc] =
          *(const v8h*)(Wt + (size_t)(k0 + 32 + rr) * ldn + n0 + cc);
#endif
    }
    const half_t* pa0 = arow0 + k0 + g * 8;
    const half_t* pa1 = arow1 + k0 + g * 8;
    v16h a0 = ld_frag(pa0, pa0 + 16);
    v16h a1 = ld_frag(pa1, pa1 + 16);
    const half_t* bp = &bsh[cur][lr];
    #pragma unroll
    for (int t = 0; t < 4; ++t) {
      v16h b = ld_col(bp + t * 16, BROW_, g);
      acc[0][t] = WMMA_F16(a0, b, acc[0][t]);
      acc[1][t] = WMMA_F16(a1, b, acc[1][t]);
    }
#if USE_TDM
    if (wv == 0 && k0 + 32 < C_) __builtin_amdgcn_s_wait_tensorcnt(0);
#endif
    __syncthreads();
  }

  #pragma unroll
  for (int t = 0; t < 4; ++t) {
    int n = n0 + t * 16 + lr;
    float bv = bias[n];
    #pragma unroll
    for (int ms = 0; ms < 2; ++ms) {
      #pragma unroll
      for (int r = 0; r < 8; ++r) {
        int m = m0 + ms * 16 + 8 * g + r;
        out[(size_t)m * C_ + n] = acc[ms][t][r] + bv;
      }
    }
  }
}

// ---------------- host launch ----------------

extern "C" void kernel_launch(void* const* d_in, const int* in_sizes, int n_in,
                              void* d_out, int out_size, void* d_ws, size_t ws_size,
                              hipStream_t stream) {
  const float* x      = (const float*)d_in[0];
  const float* w_attn = (const float*)d_in[1];
  const float* b_attn = (const float*)d_in[2];
  const float* w_proj = (const float*)d_in[3];
  const float* b_proj = (const float*)d_in[4];

  char* ws = (char*)d_ws;
  unsigned int* scales = (unsigned int*)ws;          // [0]=attn absmax bits, [1]=proj
  size_t off = 256;
  half_t* x_h  = (half_t*)(ws + off); off += (size_t)BT_ * C_ * sizeof(half_t);
  half_t* wt_a = (half_t*)(ws + off); off += (size_t)C_ * 3 * C_ * sizeof(half_t);
  half_t* wt_p = (half_t*)(ws + off); off += (size_t)C_ * C_ * sizeof(half_t);
  half_t* Qh   = (half_t*)(ws + off); off += (size_t)B_ * H_ * T_ * D_ * sizeof(half_t);
  half_t* Ktr  = (half_t*)(ws + off); off += (size_t)B_ * H_ * T_ * D_ * sizeof(half_t);
  half_t* Vh   = (half_t*)(ws + off); off += (size_t)B_ * H_ * T_ * D_ * sizeof(half_t);
  half_t* Yh   = (half_t*)(ws + off); off += (size_t)BT_ * C_ * sizeof(half_t);

  (void)hipMemsetAsync(scales, 0, 256, stream);

  absmax_kernel<<<256, 256, 0, stream>>>(w_attn, 3 * C_ * C_, scales + 0);
  absmax_kernel<<<256, 256, 0, stream>>>(w_proj, C_ * C_, scales + 1);

  snap_transpose_kernel<<<(3 * C_ * C_ + 255) / 256, 256, 0, stream>>>(
      w_attn, scales + 0, wt_a, 3 * C_, C_);
  snap_transpose_kernel<<<(C_ * C_ + 255) / 256, 256, 0, stream>>>(
      w_proj, scales + 1, wt_p, C_, C_);

  cvt_half_kernel<<<(BT_ * C_ + 255) / 256, 256, 0, stream>>>(x, x_h, BT_ * C_);

  qkv_gemm_kernel<<<dim3(BT_ / 256, (3 * C_) / 64), 256, 0, stream>>>(
      x_h, wt_a, b_attn, Qh, Ktr, Vh);

  attn_kernel<<<(B_ * H_ * (T_ / 16)) / 8, 256, 0, stream>>>(Qh, Ktr, Vh, Yh);

  proj_gemm_kernel<<<dim3(BT_ / 256, C_ / 64), 256, 0, stream>>>(
      Yh, wt_p, b_proj, (float*)d_out);
}
